// MultiHeadAttention_5480378270114
// MI455X (gfx1250) — matrix operand order repaired and checked by reading, not compiled, not run on hardware
//
#include <hip/hip_runtime.h>

typedef __bf16 bf16_t;
typedef __attribute__((ext_vector_type(16))) __bf16 v16bf;
typedef __attribute__((ext_vector_type(8)))  __bf16 v8bf;
typedef __attribute__((ext_vector_type(8)))  float  v8f;
typedef int v4i_vec __attribute__((__vector_size__(16)));

#define BB 4
#define TT 2048
#define DD 512
#define HH 8
#define HD 64

#if defined(__has_builtin)
#if __has_builtin(__builtin_amdgcn_global_load_async_to_lds_b128)
#define USE_ASYNC_LDS 1
#endif
#endif

__device__ __forceinline__ v8f wmma_bf16(v16bf a, v16bf b, v8f c) {
  // D = A(16x32 bf16) * B(32x16 bf16) + C(16x16 f32)
  return __builtin_amdgcn_wmma_f32_16x16x32_bf16(false, a, false, b, (short)0, c,
                                                 false, false);
}

union FragBF { v16bf v; v8bf h[2]; };

// A-matrix 16x32 bf16 fragment from row-major src (ISA 7.12.2):
// lane L (half hf=L>>4): row = r0 + (L&15); elems 0..7 = K {8hf..8hf+7},
// elems 8..15 = K {16+8hf..16+8hf+7}  -> two contiguous 16B loads.
__device__ __forceinline__ v16bf load_a_frag(const bf16_t* src, int ld, int r0,
                                             int k0, int lane) {
  const int r  = r0 + (lane & 15);
  const int hf = (lane >> 4) & 1;
  FragBF u;
  const bf16_t* p = src + (size_t)r * ld + k0 + 8 * hf;
  u.h[0] = *(const v8bf*)(p);
  u.h[1] = *(const v8bf*)(p + 16);
  return u.v;
}

// B-matrix 32x16 bf16 fragment where B[k][n] = S[n][k], S row-major (rows = n):
// lane L: col n = n0 + (L&15); elems 0..7 = K {8hf..8hf+7}, elems 8..15 = K {16+8hf..16+8hf+7}.
__device__ __forceinline__ v16bf load_b_frag_nmajor(const bf16_t* S, int ld,
                                                    int n0, int k0, int lane) {
  const int n  = n0 + (lane & 15);
  const int hf = (lane >> 4) & 1;
  FragBF u;
  const bf16_t* p = S + (size_t)n * ld + k0 + 8 * hf;
  u.h[0] = *(const v8bf*)(p);
  u.h[1] = *(const v8bf*)(p + 16);
  return u.v;
}

// 16-byte global -> LDS staging chunk: async path when the builtin exists.
__device__ __forceinline__ void stage16(const bf16_t* g, bf16_t* l) {
#ifdef USE_ASYNC_LDS
  __builtin_amdgcn_global_load_async_to_lds_b128(
      (__attribute__((address_space(1))) v4i_vec*)(g),
      (__attribute__((address_space(3))) v4i_vec*)(l), 0, 0);
#else
  *(v8bf*)l = *(const v8bf*)g;
#endif
}

__device__ __forceinline__ void stage_wait() {
#ifdef USE_ASYNC_LDS
#if __has_builtin(__builtin_amdgcn_s_wait_asynccnt)
  __builtin_amdgcn_s_wait_asynccnt(0);
#else
  asm volatile("s_wait_asynccnt 0x0" ::: "memory");
#endif
#endif
}

// ---------------- Kernel 1: f32 -> bf16 convert (X and W) ----------------
__global__ __launch_bounds__(256) void cvt_bf16_kernel(
    const float* __restrict__ X, const float* __restrict__ W,
    bf16_t* __restrict__ Xb, bf16_t* __restrict__ Wb) {
  const size_t NX = (size_t)BB * TT * DD;
  const size_t NW = (size_t)3 * DD * DD;
  size_t i = (size_t)blockIdx.x * 256 + threadIdx.x;
  if (i < NX) {
    Xb[i] = (bf16_t)X[i];
  } else if (i < NX + NW) {
    Wb[i - NX] = (bf16_t)W[i - NX];
  }
}

// ---------------- Kernel 2: QKV GEMM, one wave computes a 64x64 tile --------
// C[m][n] = sum_k X[m][k] * W[n][k] + bias[n]; q cols scaled by hd^-0.5.
// Q,K written bf16 row-major [t][512]; V written transposed Vt[(b*H+h)*64+d][t].
__global__ __launch_bounds__(32) void qkv_gemm_kernel(
    const bf16_t* __restrict__ X, const bf16_t* __restrict__ W,
    const float* __restrict__ bias,
    bf16_t* __restrict__ Qb, bf16_t* __restrict__ Kb, bf16_t* __restrict__ Vt) {
  const int lane = threadIdx.x & 31;
  const int nl = lane & 15;
  const int hf = lane >> 4;
  const int NT = (3 * DD) / 64;                // 24 N tiles
  const int m0 = (blockIdx.x / NT) * 64;
  const int n0 = (blockIdx.x % NT) * 64;

  v8f acc[4][4];
#pragma unroll
  for (int mi = 0; mi < 4; mi++)
#pragma unroll
    for (int ni = 0; ni < 4; ni++)
#pragma unroll
      for (int i = 0; i < 8; i++) acc[mi][ni][i] = 0.f;

  for (int k0 = 0; k0 < DD; k0 += 32) {
    if (k0 + 32 < DD) {  // hint next K-slab into cache (global_prefetch_b8)
      __builtin_prefetch(X + (size_t)(m0 + nl) * DD + k0 + 32, 0, 1);
      __builtin_prefetch(W + (size_t)(n0 + nl) * DD + k0 + 32, 0, 1);
    }
    v16bf bfr[4];
#pragma unroll
    for (int ni = 0; ni < 4; ni++)
      bfr[ni] = load_b_frag_nmajor(W, DD, n0 + 16 * ni, k0, lane);
#pragma unroll
    for (int mi = 0; mi < 4; mi++) {
      v16bf a = load_a_frag(X, DD, m0 + 16 * mi, k0, lane);
#pragma unroll
      for (int ni = 0; ni < 4; ni++)
        acc[mi][ni] = wmma_bf16(a, bfr[ni], acc[mi][ni]);
    }
  }

  const float scaling = 0.125f;                // hd^-0.5, hd = 64
#pragma unroll
  for (int mi = 0; mi < 4; mi++) {
#pragma unroll
    for (int ni = 0; ni < 4; ni++) {
      const int nb = n0 + 16 * ni;             // 16-wide subtile, uniform branch
      const int n  = nb + nl;                  // global output column (per lane)
      const float bv = bias[n];
      const int rbase = m0 + 16 * mi + 8 * hf; // first of 8 consecutive rows
      if (nb < DD) {                           // ---- Q (scaled) ----
#pragma unroll
        for (int v = 0; v < 8; v++)
          Qb[(size_t)(rbase + v) * DD + n] = (bf16_t)((acc[mi][ni][v] + bv) * scaling);
      } else if (nb < 2 * DD) {                // ---- K ----
#pragma unroll
        for (int v = 0; v < 8; v++)
          Kb[(size_t)(rbase + v) * DD + (n - DD)] = (bf16_t)(acc[mi][ni][v] + bv);
      } else {                                 // ---- V, transposed store ----
        const int n2 = n - 2 * DD;
        const int hh = n2 >> 6;
        const int d  = n2 & (HD - 1);
        const int b  = rbase >> 11;            // rbase / TT
        const int t  = rbase & (TT - 1);
        v8bf pk;
#pragma unroll
        for (int v = 0; v < 8; v++) pk[v] = (bf16_t)(acc[mi][ni][v] + bv);
        *(v8bf*)(Vt + ((size_t)(b * HH + hh) * HD + d) * TT + t) = pk;
      }
    }
  }
}

// ---------------- Kernel 3: flash attention ---------------------------------
// 4 waves per workgroup; each wave owns 16 queries (64/WG). K/V 32-key tiles
// are staged once per WG into LDS (async-to-LDS when available) and shared.
__global__ __launch_bounds__(128) void flash_attn_kernel(
    const bf16_t* __restrict__ Qb, const bf16_t* __restrict__ Kb,
    const bf16_t* __restrict__ Vt, const unsigned char* __restrict__ kmask,
    float* __restrict__ Ob) {
  const int tid = threadIdx.x;
  const int lane = tid & 31;
  const int wv = tid >> 5;                     // wave 0..3
  const int nl = lane & 15;
  const int hf = lane >> 4;
  const int wid = blockIdx.x;
  const int qt = wid & (TT / 64 - 1);          // 32 query tiles of 64
  const int bh = wid >> 5;
  const int h = bh & (HH - 1);
  const int b = bh >> 3;
  const int q0 = qt * 64 + wv * 16;            // this wave's 16 queries

  const bf16_t* Qp = Qb + (size_t)b * TT * DD + h * HD;
  const bf16_t* Kp = Kb + (size_t)b * TT * DD + h * HD;
  const bf16_t* Vp = Vt + (size_t)(b * HH + h) * HD * TT;
  const unsigned char* mp = kmask + (size_t)b * TT;

  __shared__ bf16_t Kl[32 * 64];               // K tile:  [key_local][d]
  __shared__ bf16_t Vl[64 * 32];               // Vt tile: [d][key_local]
  __shared__ bf16_t Pl[4][16 * 32];            // per-wave probs tile

  // Q fragments stay resident: hd=64 -> two Kdim=32 A-frags.
  v16bf qa0 = load_a_frag(Qp, DD, q0, 0, lane);
  v16bf qa1 = load_a_frag(Qp, DD, q0, 32, lane);

  v8f o0, o1, o2, o3;
#pragma unroll
  for (int i = 0; i < 8; i++) { o0[i] = 0.f; o1[i] = 0.f; o2[i] = 0.f; o3[i] = 0.f; }
  float m_i[8], l_i[8];
#pragma unroll
  for (int v = 0; v < 8; v++) { m_i[v] = -1e30f; l_i[v] = 0.f; }

  const float L2E = 1.4426950408889634f;

  for (int j0 = 0; j0 < TT; j0 += 32) {
    // ---- cooperative staging: 512 x 16B chunks (K: 4KB, Vt: 4KB) ----
#pragma unroll
    for (int c = 0; c < 4; c++) {
      const int i = tid + 128 * c;
      if (i < 256) {                           // K tile chunk
        const int row = i >> 3, col = (i & 7) * 8;
        stage16(Kp + (size_t)(j0 + row) * DD + col, &Kl[row * 64 + col]);
      } else {                                 // V tile chunk
        const int i2 = i - 256;
        const int d = i2 >> 2, col = (i2 & 3) * 8;
        stage16(Vp + (size_t)d * TT + j0 + col, &Vl[d * 32 + col]);
      }
    }
    stage_wait();
    __syncthreads();

    // ---- S = Q * K^T from LDS ----
    v8f s0, s1;
#pragma unroll
    for (int i = 0; i < 8; i++) { s0[i] = 0.f; s1[i] = 0.f; }
    s0 = wmma_bf16(qa0, load_b_frag_nmajor(Kl, 64, 0, 0, lane), s0);
    s0 = wmma_bf16(qa1, load_b_frag_nmajor(Kl, 64, 0, 32, lane), s0);
    s1 = wmma_bf16(qa0, load_b_frag_nmajor(Kl, 64, 16, 0, lane), s1);
    s1 = wmma_bf16(qa1, load_b_frag_nmajor(Kl, 64, 16, 32, lane), s1);

    // key padding mask: this lane's column is key j0+nl (s0) / j0+16+nl (s1).
    const bool mk0 = mp[j0 + nl] != 0;
    const bool mk1 = mp[j0 + 16 + nl] != 0;
#pragma unroll
    for (int v = 0; v < 8; v++) {              // selects: no EXEC divergence
      s0[v] = mk0 ? -1e30f : s0[v];
      s1[v] = mk1 ? -1e30f : s1[v];
    }

    float mnew[8], alpha[8];
#pragma unroll
    for (int v = 0; v < 8; v++) {              // row max over 16 column-lanes
      float t = fmaxf(s0[v], s1[v]);
      t = fmaxf(t, __shfl_xor(t, 1, 32));
      t = fmaxf(t, __shfl_xor(t, 2, 32));
      t = fmaxf(t, __shfl_xor(t, 4, 32));
      t = fmaxf(t, __shfl_xor(t, 8, 32));
      mnew[v] = fmaxf(m_i[v], t);
      alpha[v] = exp2f((m_i[v] - mnew[v]) * L2E);
      m_i[v] = mnew[v];
    }

#pragma unroll
    for (int v = 0; v < 8; v++) {
      float p0 = exp2f((s0[v] - mnew[v]) * L2E);
      float p1 = exp2f((s1[v] - mnew[v]) * L2E);
      float rs = p0 + p1;                      // row-sum across 16 column-lanes
      rs += __shfl_xor(rs, 1, 32);
      rs += __shfl_xor(rs, 2, 32);
      rs += __shfl_xor(rs, 4, 32);
      rs += __shfl_xor(rs, 8, 32);
      l_i[v] = l_i[v] * alpha[v] + rs;
      o0[v] *= alpha[v]; o1[v] *= alpha[v]; o2[v] *= alpha[v]; o3[v] *= alpha[v];
      const int row = v + 8 * hf;
      Pl[wv][row * 32 + nl]      = (bf16_t)p0;  // wave-private: in-order DS
      Pl[wv][row * 32 + nl + 16] = (bf16_t)p1;
    }

    // ---- O += P(16x32) * V(32x64): one A-frag covers all 32 keys ----
    v16bf pa = load_a_frag(Pl[wv], 32, 0, 0, lane);
    o0 = wmma_bf16(pa, load_b_frag_nmajor(Vl, 32, 0,  0, lane), o0);
    o1 = wmma_bf16(pa, load_b_frag_nmajor(Vl, 32, 16, 0, lane), o1);
    o2 = wmma_bf16(pa, load_b_frag_nmajor(Vl, 32, 32, 0, lane), o2);
    o3 = wmma_bf16(pa, load_b_frag_nmajor(Vl, 32, 48, 0, lane), o3);
    __syncthreads();                           // all waves done with Kl/Vl
  }

#pragma unroll
  for (int v = 0; v < 8; v++) {
    const float inv = 1.0f / l_i[v];
    const int t = q0 + v + 8 * hf;
    float* orow = Ob + ((size_t)b * TT + t) * DD + h * HD + nl;
    orow[0]  = o0[v] * inv;
    orow[16] = o1[v] * inv;
    orow[32] = o2[v] * inv;
    orow[48] = o3[v] * inv;
  }
}

// ---------------- Kernel 4: residual + LayerNorm (f32) ----------------------
__global__ __launch_bounds__(128) void resid_ln_kernel(
    const float* __restrict__ Ob, const float* __restrict__ X,
    const float* __restrict__ gw, const float* __restrict__ gb,
    float* __restrict__ out) {
  const int row = blockIdx.x;
  const int tid = threadIdx.x;
  const float* xr = X + (size_t)row * DD;
  const float* orow = Ob + (size_t)row * DD;

  float vals[4];
  float s = 0.f;
#pragma unroll
  for (int i = 0; i < 4; i++) {
    const int c = tid + 128 * i;
    const float t = orow[c] + xr[c];           // residual
    vals[i] = t;
    s += t;
  }
  __shared__ float sm1[4], sm2[4];
  for (int off = 16; off > 0; off >>= 1) s += __shfl_xor(s, off, 32);
  const int wv = tid >> 5, ln = tid & 31;
  if (ln == 0) sm1[wv] = s;
  __syncthreads();
  const float mean = (sm1[0] + sm1[1] + sm1[2] + sm1[3]) * (1.0f / DD);

  float vs = 0.f;
#pragma unroll
  for (int i = 0; i < 4; i++) { const float d = vals[i] - mean; vs += d * d; }
  for (int off = 16; off > 0; off >>= 1) vs += __shfl_xor(vs, off, 32);
  if (ln == 0) sm2[wv] = vs;
  __syncthreads();
  const float var = (sm2[0] + sm2[1] + sm2[2] + sm2[3]) * (1.0f / DD);
  const float rstd = rsqrtf(var + 1e-5f);
#pragma unroll
  for (int i = 0; i < 4; i++) {
    const int c = tid + 128 * i;
    out[(size_t)row * DD + c] = (vals[i] - mean) * rstd * gw[c] + gb[c];
  }
}

extern "C" void kernel_launch(void* const* d_in, const int* in_sizes, int n_in,
                              void* d_out, int out_size, void* d_ws, size_t ws_size,
                              hipStream_t stream) {
  (void)in_sizes; (void)n_in; (void)out_size; (void)ws_size;
  const float* query = (const float*)d_in[0];
  const unsigned char* kmask = (const unsigned char*)d_in[1];   // bool mask
  const float* W     = (const float*)d_in[2];
  const float* bias  = (const float*)d_in[3];
  const float* lnw   = (const float*)d_in[4];
  const float* lnb   = (const float*)d_in[5];
  float* out = (float*)d_out;

  // Workspace carve-out (~52 MB)
  constexpr size_t NX  = (size_t)BB * TT * DD;       // 4,194,304
  constexpr size_t NW  = (size_t)3 * DD * DD;        //   786,432
  char* ws = (char*)d_ws;
  size_t off = 0;
  bf16_t* Xbf = (bf16_t*)(ws + off); off += NX * sizeof(bf16_t);
  bf16_t* Wbf = (bf16_t*)(ws + off); off += NW * sizeof(bf16_t);
  bf16_t* Qb  = (bf16_t*)(ws + off); off += NX * sizeof(bf16_t);
  bf16_t* Kb  = (bf16_t*)(ws + off); off += NX * sizeof(bf16_t);
  bf16_t* Vt  = (bf16_t*)(ws + off); off += NX * sizeof(bf16_t);
  float*  Ob  = (float*) (ws + off); off += NX * sizeof(float);

  // 1) f32 -> bf16
  {
    const size_t n = NX + NW;
    cvt_bf16_kernel<<<(unsigned)((n + 255) / 256), 256, 0, stream>>>(query, W, Xbf, Wbf);
  }
  // 2) fused QKV projection (WMMA bf16), 64x64 tiles, bias + q-scaling fused
  {
    const int grid = (BB * TT / 64) * (3 * DD / 64);  // 128 * 24 = 3072
    qkv_gemm_kernel<<<grid, 32, 0, stream>>>(Xbf, Wbf, bias, Qb, Kb, Vt);
  }
  // 3) flash attention (WMMA bf16, online softmax, LDS-shared K/V tiles)
  {
    const int grid = BB * HH * (TT / 64);             // 1024
    flash_attn_kernel<<<grid, 128, 0, stream>>>(Qb, Kb, Vt, kmask, Ob);
  }
  // 4) residual + LayerNorm
  resid_ln_kernel<<<BB * TT, 128, 0, stream>>>(Ob, query, lnw, lnb, out);
}
